// TypeDP_14731737825577
// MI455X (gfx1250) — compile-verified, matrix operand order checked by reading, not run on hardware
//
#include <hip/hip_runtime.h>
#include <hip/hip_bf16.h>

// ---------- types / WMMA helpers -------------------------------------------
typedef __attribute__((ext_vector_type(16))) _Float16 v16h;
typedef __attribute__((ext_vector_type(8)))  _Float16 v8h;
typedef __attribute__((ext_vector_type(8)))  float    v8f;

__device__ __forceinline__ v8f wmma_f16(v16h a, v16h b, v8f c) {
  // D = A(16x32 f16) * B(32x16 f16) + C(16x16 f32)
  return __builtin_amdgcn_wmma_f32_16x16x32_f16(
      /*neg_a=*/false, a, /*neg_b=*/false, b,
      /*c_mod=*/(short)0, c, /*reuse_a=*/false, /*reuse_b=*/false);
}

// Branch-free tanh on the hardware transcendental pipe:
// tanh(x) = sign(x) * (1 - e^{-2|x|}) / (1 + e^{-2|x|})
// -> v_exp_f32 + v_rcp_f32, no EXEC divergence; TRANS32 co-executes with WMMA.
__device__ __forceinline__ float fast_tanh(float x) {
  float ax = __builtin_fabsf(x);
  float e  = __builtin_amdgcn_exp2f(-2.8853900817779268f * ax);  // e^{-2|x|}
  float t  = (1.0f - e) * __builtin_amdgcn_rcpf(1.0f + e);
  return __builtin_copysignf(t, x);
}

// Load a 16(row) x 32(K) f16 fragment from a row-major [rows][ld] array.
// Per ISA 16-bit A layout: lanes 0-15 hold row=lane, K 0..7 (v0-3) and
// 16..23 (v4-7); lanes 16-31 hold row=lane-16, K 8..15 and 24..31.
// B fragments use the same loader on the transposed (N-major) weight array.
__device__ __forceinline__ v16h load_frag(const _Float16* base, int ld,
                                          int row0, int k0) {
  const int lane = threadIdx.x & 31;
  const int l = lane & 15, hi = lane >> 4;
  const _Float16* p = base + (size_t)(row0 + l) * ld + k0 + hi * 8;
  v8h lo = *(const v8h*)(p);
  v8h hh = *(const v8h*)(p + 16);
  v16h r;
#pragma unroll
  for (int i = 0; i < 8; ++i) { r[i] = lo[i]; r[8 + i] = hh[i]; }
  return r;
}

// ---------- problem constants ----------------------------------------------
#define NATOMS   1024
#define BATCH    4
#define JDIM     200      // NTYPES * MAX_NEIGH
#define JCH      112      // padded chunk (100 real neighbors per type)
#define FIT_IN   1600     // M2 * 100
#define FIT_H    240
#define FIT_KP   256      // FIT_H padded to K multiple of 32

// ---------- workspace layout (bytes) ---------------------------------------
#define WS_DR    ((size_t)0)                              // f16 [4096][1600]
#define WS_W0T   (WS_DR  + (size_t)4096 * FIT_IN * 2)     // f16 [2][240][1600]
#define WS_W1T   (WS_W0T + (size_t)2 * FIT_H * FIT_IN * 2)// f16 [2][240][256]
#define WS_W2T   (WS_W1T + (size_t)2 * FIT_H * FIT_KP * 2)// f16 [2][240][256]

// ============================================================================
// Prep: transpose fit weights to f16 [N][K] (K zero-padded), zero Etot slots.
// ============================================================================
__global__ void prep_kernel(const float* __restrict__ W0,
                            const float* __restrict__ W1,
                            const float* __restrict__ W2,
                            _Float16* __restrict__ W0t,
                            _Float16* __restrict__ W1t,
                            _Float16* __restrict__ W2t,
                            float* __restrict__ out) {
  int i = blockIdx.x * 256 + threadIdx.x;
  if (i < 4) out[i] = 0.0f;                       // Etot accumulators
  if (i < 2 * FIT_H * FIT_IN) {                   // W0t[t][n][k] = W0[t][k][n]
    int t = i / (FIT_H * FIT_IN);
    int rem = i % (FIT_H * FIT_IN);
    int n = rem / FIT_IN, k = rem % FIT_IN;
    W0t[i] = (_Float16)W0[(size_t)t * FIT_IN * FIT_H + (size_t)k * FIT_H + n];
  }
  if (i < 2 * FIT_H * FIT_KP) {                   // pad K 240..255 with zeros
    int t = i / (FIT_H * FIT_KP);
    int rem = i % (FIT_H * FIT_KP);
    int n = rem / FIT_KP, k = rem % FIT_KP;
    float v1 = (k < FIT_H) ? W1[(size_t)t * FIT_H * FIT_H + k * FIT_H + n] : 0.f;
    float v2 = (k < FIT_H) ? W2[(size_t)t * FIT_H * FIT_H + k * FIT_H + n] : 0.f;
    W1t[i] = (_Float16)v1;
    W2t[i] = (_Float16)v2;
  }
}

// ============================================================================
// Embedding kernel: one workgroup (256 threads = 8 waves) per (b,n) atom.
// Layers 2/3 of the per-neighbor MLP run as f16 WMMA GEMMs over neighbors.
// ============================================================================
__global__ void embed_kernel(const float* __restrict__ Ri,
                             const float* __restrict__ tv,
                             const float* __restrict__ W1,
                             const float* __restrict__ b1,
                             const float* __restrict__ W2,
                             const float* __restrict__ b2,
                             const float* __restrict__ W3,
                             const float* __restrict__ b3,
                             _Float16* __restrict__ DRout) {
  extern __shared__ char smem[];
  float*    sRi  = (float*)(smem);                  // [200][4]      3200 B
  float*    sXyz = (float*)(smem + 3200);           // [4][112]      1792 B
  float*    sW1  = (float*)(smem + 4992);           // [32]
  float*    sB1  = sW1 + 32;                        // [32]
  float*    sB2  = sB1 + 32;                        // [64]
  float*    sB3  = sB2 + 64;                        // [112]   (cum 5952)
  _Float16* sH1  = (_Float16*)(smem + 5952);        // [112][32]     7168 B
  _Float16* sH2  = (_Float16*)(smem + 13120);       // [112][64]    14336 B
  _Float16* sW2t = (_Float16*)(smem + 27456);       // [64][32]      4096 B
  _Float16* sW3t = (_Float16*)(smem + 31552);       // [112][64]    14336 B
  float*    sG   = (float*)(smem + 45888);          // [112][112]   50176 B
  // total 96064 B dynamic LDS

  const int atom = blockIdx.x;                      // 0..4095
  const int tid  = threadIdx.x;
  const int wave = tid >> 5;
  const int lane = tid & 31;
  const int l    = lane & 15, hi = lane >> 4;

  const float* ri = Ri + (size_t)atom * JDIM * 4;
  for (int i = tid; i < JDIM * 4; i += 256) sRi[i] = ri[i];
  for (int i = tid; i < 4 * JCH; i += 256) sXyz[i] = 0.f;
  // zero the K-pad columns (50..63) of sH2 exactly once; post-ops never
  // write them again, so the layer-3 WMMA always sees zero padding.
  for (int i = tid; i < JCH * 14; i += 256) {
    int j = i / 14, cc = 50 + (i % 14);
    sH2[j * 64 + cc] = (_Float16)0.f;
  }
  __syncthreads();

  for (int chunk = 0; chunk < 2; ++chunk) {
    for (int i = tid; i < JCH * JCH; i += 256) sG[i] = 0.f;
    __syncthreads();

    for (int k = 0; k < 5; ++k) {
      // ---- stage per-prop weights (transposed, zero-padded) into LDS ----
      for (int i = tid; i < 32; i += 256) {
        sW1[i] = (i < 25) ? W1[k * 25 + i] : 0.f;
        sB1[i] = (i < 25) ? b1[k * 25 + i] : 0.f;
      }
      for (int i = tid; i < 64; i += 256) sB2[i] = (i < 50) ? b2[k * 50 + i] : 0.f;
      for (int i = tid; i < JCH; i += 256) sB3[i] = (i < 100) ? b3[k * 100 + i] : 0.f;
      for (int i = tid; i < 64 * 32; i += 256) {
        int o = i >> 5, kk = i & 31;                 // W2t[o][kk] = W2[k][kk][o]
        sW2t[i] = (o < 50 && kk < 25) ? (_Float16)W2[k * 1250 + kk * 50 + o]
                                      : (_Float16)0.f;
      }
      for (int i = tid; i < JCH * 64; i += 256) {
        int o = i >> 6, kk = i & 63;                 // W3t[o][kk] = W3[k][kk][o]
        sW3t[i] = (o < 100 && kk < 50) ? (_Float16)W3[k * 5000 + kk * 100 + o]
                                       : (_Float16)0.f;
      }
      __syncthreads();

      // ---- layer 1 (elementwise): h1 = tanh(s*w1 + b1), K-pad cols = 0 ----
      for (int i = tid; i < JCH * 32; i += 256) {
        int j = i >> 5, c = i & 31;
        float s = (j < 100) ? sRi[(chunk * 100 + j) * 4 + 0] : 0.f;
        sH1[i] = (_Float16)fast_tanh(s * sW1[c] + sB1[c]);  // pad c: tanh(0)=0
      }
      __syncthreads();

      // ---- layer 2 WMMA: (112x32) x (32x64), 7x4 tiles ----
      for (int it = 0; it < 4; ++it) {
        int p = wave + it * 8;
        if (p < 28) {
          int mt = p >> 2, nt = p & 3;
          v16h a  = load_frag(sH1, 32, mt * 16, 0);
          v16h bf = load_frag(sW2t, 32, nt * 16, 0);
          v8f acc = {};
          acc = wmma_f16(a, bf, acc);
          const int c = nt * 16 + l;
          if (c < 50) {                              // lane-invariant over e
            const float bias = sB2[c];
            const _Float16* res = sH1 + (c % 25);    // residual column
            _Float16* outp = sH2 + c;
            const int jbase = mt * 16 + hi * 8;
#pragma unroll
            for (int e = 0; e < 8; ++e) {
              int j = jbase + e;
              float v = fast_tanh(acc[e] + bias) + (float)res[j * 32];
              outp[j * 64] = (_Float16)v;
            }
          }
        }
      }
      __syncthreads();

      // ---- layer 3 WMMA: (112x64) x (64x112), 7x7 tiles, 2 K-steps ----
      const float tvk = tv[chunk * 5 + k];
      for (int it = 0; it < 7; ++it) {
        int p = wave + it * 8;
        if (p < 49) {
          int mt = p / 7, nt = p % 7;
          v8f acc = {};
#pragma unroll
          for (int ks = 0; ks < 2; ++ks) {
            v16h a  = load_frag(sH2, 64, mt * 16, ks * 32);
            v16h bf = load_frag(sW3t, 64, nt * 16, ks * 32);
            acc = wmma_f16(a, bf, acc);
          }
          const int c = nt * 16 + l;
          if (c < 100) {                             // lane-invariant over e
            const float bias = sB3[c];
            const _Float16* res = sH2 + (c % 50);    // residual column
            float* gp = sG + c;
            const int jbase = mt * 16 + hi * 8;
#pragma unroll
            for (int e = 0; e < 8; ++e) {
              int j = jbase + e;
              float v = fast_tanh(acc[e] + bias) + (float)res[j * 64];
              gp[j * JCH] += tvk * v;                // prop-weighted sum
            }
          }
        }
      }
      __syncthreads();
    } // props

    // ---- xyz[c][d] += sum_j Ri[j][c] * G[j][d] ----
    for (int i = tid; i < 4 * 100; i += 256) {
      int c = i / 100, d = i % 100;
      float acc = 0.f;
      for (int j = 0; j < 100; ++j)
        acc += sRi[(chunk * 100 + j) * 4 + c] * sG[j * JCH + d];
      sXyz[c * JCH + d] += acc;
    }
    __syncthreads();
  } // chunks

  for (int i = tid; i < 4 * JCH; i += 256) sXyz[i] *= (1.0f / 200.0f);
  __syncthreads();

  // ---- DR[i][jj] = sum_c xyz[c][i]*xyz[c][jj], write f16 row of 1600 ----
  _Float16* dr = DRout + (size_t)atom * FIT_IN;
  for (int i = tid; i < FIT_IN; i += 256) {
    int ii = i >> 4, jj = i & 15;
    float acc = 0.f;
#pragma unroll
    for (int c = 0; c < 4; ++c)
      acc += sXyz[c * JCH + ii] * sXyz[c * JCH + jj];
    dr[i] = (_Float16)acc;
  }
}

// ============================================================================
// Fitting kernel: one workgroup per (b, t, 64-row block). 3 WMMA GEMM layers.
// ============================================================================
__global__ void fit_kernel(const _Float16* __restrict__ DRf,
                           const _Float16* __restrict__ W0t,
                           const _Float16* __restrict__ W1t,
                           const _Float16* __restrict__ W2t,
                           const float* __restrict__ b0,
                           const float* __restrict__ b1,
                           const float* __restrict__ b2,
                           const float* __restrict__ W3,
                           const float* __restrict__ b3,
                           float* __restrict__ out) {
  extern __shared__ char smem[];
  _Float16* sA   = (_Float16*)(smem);               // [64][32]     4096 B
  _Float16* sB   = sA + 64 * 32;                    // [240][32]   15360 B
  _Float16* sH0  = sB + 240 * 32;                   // [64][256]   32768 B
  _Float16* sH1  = sH0 + 64 * FIT_KP;               // [64][256]   32768 B
  float*    sBias = (float*)(sH1 + 64 * FIT_KP);    // [240]
  float*    sW3   = sBias + FIT_H;                  // [240]
  // total 86912 B dynamic LDS

  const int blk = blockIdx.x;                       // 0..63
  const int mb = blk & 7, t = (blk >> 3) & 1, b = blk >> 4;
  const int r0 = b * NATOMS + t * 512 + mb * 64;
  const int tid = threadIdx.x, wave = tid >> 5;
  const int lane = tid & 31, l = lane & 15, hi = lane >> 4;

  // zero K-pad columns of the activation buffers once
  for (int i = tid; i < 64 * 16; i += 256) {
    int rr = i >> 4, cc = FIT_H + (i & 15);
    sH0[rr * FIT_KP + cc] = (_Float16)0.f;
    sH1[rr * FIT_KP + cc] = (_Float16)0.f;
  }
  for (int i = tid; i < FIT_H; i += 256) sW3[i] = W3[t * FIT_H + i];

  v8f acc[8];
  const _Float16* A0 = DRf + (size_t)r0 * FIT_IN;

  // -------- layer 0: h0 = tanh(DR @ W0 + b0), K=1600 --------
#pragma unroll
  for (int i = 0; i < 8; ++i) acc[i] = (v8f){};
  {
    const _Float16* B0 = W0t + (size_t)t * FIT_H * FIT_IN;
    for (int ks = 0; ks < FIT_IN / 32; ++ks) {
      for (int i = tid; i < 256; i += 256) {        // 64x32 A tile, v8h loads
        int rr = i >> 2, cc = (i & 3) * 8;
        *(v8h*)(sA + rr * 32 + cc) =
            *(const v8h*)(A0 + (size_t)rr * FIT_IN + ks * 32 + cc);
      }
      for (int i = tid; i < 960; i += 256) {        // 240x32 B tile
        int rr = i >> 2, cc = (i & 3) * 8;
        *(v8h*)(sB + rr * 32 + cc) =
            *(const v8h*)(B0 + (size_t)rr * FIT_IN + ks * 32 + cc);
        if (ks + 1 < FIT_IN / 32)                   // pull next K tile to L2/L0
          __builtin_prefetch(B0 + (size_t)rr * FIT_IN + (ks + 1) * 32 + cc, 0, 1);
      }
      __syncthreads();
#pragma unroll
      for (int i = 0; i < 8; ++i) {
        int p = wave + i * 8;
        if (p < 60) {
          int mt = p / 15, nt = p % 15;
          v16h a  = load_frag(sA, 32, mt * 16, 0);
          v16h bf = load_frag(sB, 32, nt * 16, 0);
          acc[i] = wmma_f16(a, bf, acc[i]);
        }
      }
      __syncthreads();
    }
    for (int i = tid; i < FIT_H; i += 256) sBias[i] = b0[t * FIT_H + i];
    __syncthreads();
#pragma unroll
    for (int i = 0; i < 8; ++i) {
      int p = wave + i * 8;
      if (p < 60) {
        int mt = p / 15, nt = p % 15;
        const int n = nt * 16 + l;
        const float bias = sBias[n];
        const int mbase = mt * 16 + hi * 8;
#pragma unroll
        for (int e = 0; e < 8; ++e)
          sH0[(mbase + e) * FIT_KP + n] = (_Float16)fast_tanh(acc[i][e] + bias);
      }
    }
    __syncthreads();
  }

  // -------- layers 1 and 2: h = tanh(h_prev @ W + b) + h_prev, K=256 ------
  for (int layer = 1; layer <= 2; ++layer) {
    const _Float16* Bt = (layer == 1 ? W1t : W2t) + (size_t)t * FIT_H * FIT_KP;
    const float* bb = (layer == 1 ? b1 : b2) + t * FIT_H;
    const _Float16* Hin = (layer == 1) ? sH0 : sH1;
    _Float16* Hout      = (layer == 1) ? sH1 : sH0;
#pragma unroll
    for (int i = 0; i < 8; ++i) acc[i] = (v8f){};
    for (int ks = 0; ks < FIT_KP / 32; ++ks) {
      for (int i = tid; i < 960; i += 256) {
        int rr = i >> 2, cc = (i & 3) * 8;
        *(v8h*)(sB + rr * 32 + cc) =
            *(const v8h*)(Bt + (size_t)rr * FIT_KP + ks * 32 + cc);
        if (ks + 1 < FIT_KP / 32)
          __builtin_prefetch(Bt + (size_t)rr * FIT_KP + (ks + 1) * 32 + cc, 0, 1);
      }
      __syncthreads();
#pragma unroll
      for (int i = 0; i < 8; ++i) {
        int p = wave + i * 8;
        if (p < 60) {
          int mt = p / 15, nt = p % 15;
          v16h a  = load_frag(Hin, FIT_KP, mt * 16, ks * 32);
          v16h bf = load_frag(sB, 32, nt * 16, 0);
          acc[i] = wmma_f16(a, bf, acc[i]);
        }
      }
      __syncthreads();
    }
    for (int i = tid; i < FIT_H; i += 256) sBias[i] = bb[i];
    __syncthreads();
#pragma unroll
    for (int i = 0; i < 8; ++i) {
      int p = wave + i * 8;
      if (p < 60) {
        int mt = p / 15, nt = p % 15;
        const int n = nt * 16 + l;
        const float bias = sBias[n];
        const int mbase = mt * 16 + hi * 8;
#pragma unroll
        for (int e = 0; e < 8; ++e) {
          int m = mbase + e;
          float v = fast_tanh(acc[i][e] + bias) + (float)Hin[m * FIT_KP + n];
          Hout[m * FIT_KP + n] = (_Float16)v;
        }
      }
    }
    __syncthreads();
  }

  // -------- output layer: Ei = h2 . W3 + b3; Etot via atomics --------
  if (tid < 64) {
    float accd = 0.f;
    for (int c = 0; c < FIT_H; c += 8) {
      v8h hv = *(const v8h*)(sH0 + tid * FIT_KP + c);  // h2 lives in sH0
#pragma unroll
      for (int q = 0; q < 8; ++q) accd += (float)hv[q] * sW3[c + q];
    }
    float ei = accd + b3[t];
    out[4 + r0 + tid] = ei;
    atomicAdd(out + b, ei);
  }
}

// ============================================================================
extern "C" void kernel_launch(void* const* d_in, const int* in_sizes, int n_in,
                              void* d_out, int out_size, void* d_ws,
                              size_t ws_size, hipStream_t stream) {
  const float* Ri   = (const float*)d_in[0];
  const float* tv   = (const float*)d_in[1];
  const float* eW1  = (const float*)d_in[2];
  const float* eb1  = (const float*)d_in[3];
  const float* eW2  = (const float*)d_in[4];
  const float* eb2  = (const float*)d_in[5];
  const float* eW3  = (const float*)d_in[6];
  const float* eb3  = (const float*)d_in[7];
  const float* fW0  = (const float*)d_in[8];
  const float* fb0  = (const float*)d_in[9];
  const float* fW1  = (const float*)d_in[10];
  const float* fb1  = (const float*)d_in[11];
  const float* fW2  = (const float*)d_in[12];
  const float* fb2  = (const float*)d_in[13];
  const float* fW3  = (const float*)d_in[14];
  const float* fb3  = (const float*)d_in[15];
  float* out = (float*)d_out;

  char* ws = (char*)d_ws;
  _Float16* DRf = (_Float16*)(ws + WS_DR);
  _Float16* W0t = (_Float16*)(ws + WS_W0T);
  _Float16* W1t = (_Float16*)(ws + WS_W1T);
  _Float16* W2t = (_Float16*)(ws + WS_W2T);

  prep_kernel<<<dim3((2 * FIT_H * FIT_IN + 255) / 256), dim3(256), 0, stream>>>(
      fW0, fW1, fW2, W0t, W1t, W2t, out);

  embed_kernel<<<dim3(BATCH * NATOMS), dim3(256), 96064, stream>>>(
      Ri, tv, eW1, eb1, eW2, eb2, eW3, eb3, DRf);

  fit_kernel<<<dim3(64), dim3(256), 86912, stream>>>(
      DRf, W0t, W1t, W2t, fb0, fb1, fb2, fW3, fb3, out);
}
